// ChaosStudentLM_29480655520349
// MI455X (gfx1250) — compile-verified, matrix operand order checked by reading, not run on hardware
//
#include <hip/hip_runtime.h>
#include <stdint.h>

// ---------------------------------------------------------------------------
// Types for CDNA5 WMMA (wave32): v_wmma_f32_16x16x32_bf16
// ---------------------------------------------------------------------------
typedef __bf16 bf16_t;
typedef __attribute__((ext_vector_type(16))) __bf16 v16bf;
typedef __attribute__((ext_vector_type(8)))  float  v8f;

union AFrag {
    v16bf    v;
    uint32_t u[8];
};

// fp32 -> bf16 (round-to-nearest-even) without relying on __bf16 conversion codegen
static __device__ __forceinline__ bf16_t to_bf16(float f) {
    uint32_t u = __builtin_bit_cast(uint32_t, f);
    u = (u + 0x7FFFu + ((u >> 16) & 1u)) >> 16;
    uint16_t s = (uint16_t)u;
    return __builtin_bit_cast(bf16_t, s);
}

#define DIMC      2048
#define NHEADS    16
#define HDIM      128
#define FFDIM     4096

// ---------------------------------------------------------------------------
// CDNA5 async global->LDS copy (ASYNCcnt) with sync fallback.
// Builtin signature (from hipcc diagnostic): param0 = AS(1) "int __vector(4)"*,
// param1 = LDS pointer, then two immediates (offset, cpol).
// ---------------------------------------------------------------------------
#if defined(__gfx1250__) && __has_builtin(__builtin_amdgcn_global_load_async_to_lds_b128)
#define HAVE_ASYNC 1
#else
#define HAVE_ASYNC 0
#endif

#if HAVE_ASYNC
typedef int v4i_gcc __attribute__((vector_size(16)));
typedef __attribute__((address_space(1))) v4i_gcc* gas_v4i_p;
typedef __attribute__((address_space(3))) v4i_gcc* las_v4i_p;
#endif

static __device__ __forceinline__ void async_copy16(const bf16_t* g, bf16_t* l) {
#if HAVE_ASYNC
    bf16_t* gnc = const_cast<bf16_t*>(g);
    __builtin_amdgcn_global_load_async_to_lds_b128((gas_v4i_p)gnc, (las_v4i_p)l, 0, 0);
#else
    *reinterpret_cast<int4*>(l) = *reinterpret_cast<const int4*>(g);
#endif
}

static __device__ __forceinline__ void async_wait0() {
#if HAVE_ASYNC
#if __has_builtin(__builtin_amdgcn_s_wait_asynccnt)
    __builtin_amdgcn_s_wait_asynccnt(0);
#else
    asm volatile("s_wait_asynccnt 0" ::: "memory");
#endif
#endif
}

// ---------------------------------------------------------------------------
// Weight cast + transpose:  W[K][N] f32  ->  Wt[N][K] bf16
// ---------------------------------------------------------------------------
__global__ __launch_bounds__(256)
void k_transpose_cast(const float* __restrict__ W, bf16_t* __restrict__ Wt,
                      int K, int N) {
    __shared__ float tile[32][33];
    const int kb = blockIdx.y * 32;
    const int nb = blockIdx.x * 32;
    const int tx = threadIdx.x & 31;
    const int ty = threadIdx.x >> 5;   // 0..7
#pragma unroll
    for (int i = 0; i < 32; i += 8)
        tile[ty + i][tx] = W[(size_t)(kb + ty + i) * N + nb + tx];
    __syncthreads();
#pragma unroll
    for (int i = 0; i < 32; i += 8)
        Wt[(size_t)(nb + ty + i) * K + kb + tx] = to_bf16(tile[tx][ty + i]);
}

// ---------------------------------------------------------------------------
// RMSNorm: one block per row, fp32 in, bf16 out
// ---------------------------------------------------------------------------
__global__ __launch_bounds__(256)
void k_rmsnorm_bf16(const float* __restrict__ x, const float* __restrict__ g,
                    bf16_t* __restrict__ out, int D) {
    __shared__ float red[8];
    const int row = blockIdx.x;
    const float* xr = x + (size_t)row * D;
    float ss = 0.f;
    for (int i = threadIdx.x; i < D; i += 256) { float v = xr[i]; ss += v * v; }
#pragma unroll
    for (int off = 16; off > 0; off >>= 1) ss += __shfl_down(ss, off, 32);
    if ((threadIdx.x & 31) == 0) red[threadIdx.x >> 5] = ss;
    __syncthreads();
    if (threadIdx.x < 8) {
        float v = red[threadIdx.x];
#pragma unroll
        for (int off = 4; off > 0; off >>= 1) v += __shfl_down(v, off, 8);
        if (threadIdx.x == 0) red[0] = v;
    }
    __syncthreads();
    const float r = rsqrtf(red[0] / (float)D + 1e-6f);
    bf16_t* orow = out + (size_t)row * D;
    for (int i = threadIdx.x; i < D; i += 256)
        orow[i] = to_bf16(xr[i] * r * g[i]);
}

// ---------------------------------------------------------------------------
// Tiled bf16 WMMA GEMM:  C[M][N] f32 = A[M][K] bf16 * Bt[N][K]^T bf16
// 128x128 block tile, BK=32, 256 threads = 8 waves; wave -> 32x64 strip
// (8 accumulators, 8 WMMA per K-step). Double-buffered LDS filled with
// CDNA5 async global->LDS copies; fragments load as 2x ds_load_b128.
// ---------------------------------------------------------------------------
#define GBM 128
#define GBN 128
#define GBK 32
#define LDK 48    // padded LDS K-stride (96B rows: keeps b128 16B-aligned)

__global__ __launch_bounds__(256)
void k_gemm_bf16(const bf16_t* __restrict__ A, const bf16_t* __restrict__ Bt,
                 float* __restrict__ C, int M, int N, int K) {
    __shared__ bf16_t As[2][GBM][LDK];
    __shared__ bf16_t Bs[2][GBN][LDK];

    const int tid  = threadIdx.x;
    const int bm   = blockIdx.y * GBM;
    const int bn   = blockIdx.x * GBN;
    const int wave = tid >> 5;
    const int lane = tid & 31;
    const int wm   = wave & 3;    // 4 waves over M: 32 rows each
    const int wn   = wave >> 2;   // 2 waves over N: 64 cols each
    const int ln   = lane & 15;
    const int hi   = lane >> 4;

    v8f acc[2][4];
    {
        const v8f z = {};
#pragma unroll
        for (int i = 0; i < 2; ++i)
#pragma unroll
            for (int j = 0; j < 4; ++j) acc[i][j] = z;
    }

    // staging: 256 threads x (2 rows of A + 2 rows of B) x 16B
    const int lr = tid >> 2;          // 0..63
    const int lc = (tid & 3) * 8;     // 0,8,16,24 (bf16 elems; 16B granules)

    const bf16_t* ga0 = &A [(size_t)(bm + lr)      * K + lc];
    const bf16_t* ga1 = &A [(size_t)(bm + lr + 64) * K + lc];
    const bf16_t* gb0 = &Bt[(size_t)(bn + lr)      * K + lc];
    const bf16_t* gb1 = &Bt[(size_t)(bn + lr + 64) * K + lc];

    auto stage = [&](int buf, int kt) {
        const size_t off = (size_t)kt * GBK;
        async_copy16(ga0 + off, &As[buf][lr][lc]);
        async_copy16(ga1 + off, &As[buf][lr + 64][lc]);
        async_copy16(gb0 + off, &Bs[buf][lr][lc]);
        async_copy16(gb1 + off, &Bs[buf][lr + 64][lc]);
    };

    const int ntiles = K / GBK;
    stage(0, 0);
    async_wait0();
    __syncthreads();

    for (int kt = 0; kt < ntiles; ++kt) {
        const int buf = kt & 1;
        if (kt + 1 < ntiles) stage(buf ^ 1, kt + 1);   // prefetch next tile (async)

        AFrag af[2], bf[4];
#pragma unroll
        for (int i = 0; i < 2; ++i) {
            const bf16_t* row = &As[buf][wm * 32 + i * 16 + ln][8 * hi];
            *reinterpret_cast<int4*>(&af[i].u[0]) = *reinterpret_cast<const int4*>(row);
            *reinterpret_cast<int4*>(&af[i].u[4]) = *reinterpret_cast<const int4*>(row + 16);
        }
#pragma unroll
        for (int j = 0; j < 4; ++j) {
            const bf16_t* row = &Bs[buf][wn * 64 + j * 16 + ln][8 * hi];
            *reinterpret_cast<int4*>(&bf[j].u[0]) = *reinterpret_cast<const int4*>(row);
            *reinterpret_cast<int4*>(&bf[j].u[4]) = *reinterpret_cast<const int4*>(row + 16);
        }
#pragma unroll
        for (int i = 0; i < 2; ++i)
#pragma unroll
            for (int j = 0; j < 4; ++j)
                acc[i][j] = __builtin_amdgcn_wmma_f32_16x16x32_bf16(
                    false, af[i].v, false, bf[j].v, (short)0, acc[i][j], false, false);

        async_wait0();
        __syncthreads();
    }

    // C/D layout: VGPR r -> row r + 8*hi ; col = lane&15
#pragma unroll
    for (int i = 0; i < 2; ++i) {
        const int cm = bm + wm * 32 + i * 16 + 8 * hi;
#pragma unroll
        for (int j = 0; j < 4; ++j) {
            const int cn = bn + wn * 64 + j * 16 + ln;
#pragma unroll
            for (int r = 0; r < 8; ++r)
                C[(size_t)(cm + r) * N + cn] = acc[i][j][r];
        }
    }
}

// ---------------------------------------------------------------------------
// RoPE + split: qkv f32 [B*S, 3D] -> Q,K bf16 [B,H,S,HD] (roped), Vt bf16 [B,H,HD,S]
// ---------------------------------------------------------------------------
__global__ __launch_bounds__(256)
void k_rope_split(const float* __restrict__ qkv,
                  bf16_t* __restrict__ Q, bf16_t* __restrict__ K,
                  bf16_t* __restrict__ Vt, int B, int S) {
    const int HALF = HDIM / 2;
    size_t idx = (size_t)blockIdx.x * blockDim.x + threadIdx.x;
    size_t total = (size_t)B * S * NHEADS * HALF;
    if (idx >= total) return;
    const int i = (int)(idx % HALF); idx /= HALF;
    const int h = (int)(idx % NHEADS); idx /= NHEADS;
    const int s = (int)(idx % S);
    const int b = (int)(idx / S);

    const float inv = __powf(10000.0f, -2.0f * (float)i / (float)HDIM);
    float sn, cs;
    __sincosf((float)s * inv, &sn, &cs);

    const size_t rowoff = (size_t)(b * S + s) * (size_t)(3 * DIMC);
    const int col = h * HDIM + 2 * i;
    const float qe = qkv[rowoff + col],            qo = qkv[rowoff + col + 1];
    const float ke = qkv[rowoff + DIMC + col],     ko = qkv[rowoff + DIMC + col + 1];
    const float v0 = qkv[rowoff + 2 * DIMC + col], v1 = qkv[rowoff + 2 * DIMC + col + 1];

    const int    bh = b * NHEADS + h;
    const size_t ho = (((size_t)bh) * S + s) * HDIM + 2 * i;
    Q[ho]     = to_bf16(qe * cs - qo * sn);
    Q[ho + 1] = to_bf16(qo * cs + qe * sn);
    K[ho]     = to_bf16(ke * cs - ko * sn);
    K[ho + 1] = to_bf16(ko * cs + ke * sn);
    // V transposed: [b,h,d,s]
    const size_t vbase = ((size_t)bh) * HDIM * S;
    Vt[vbase + (size_t)(2 * i)     * S + s] = to_bf16(v0);
    Vt[vbase + (size_t)(2 * i + 1) * S + s] = to_bf16(v1);
}

// ---------------------------------------------------------------------------
// Causal flash attention, bf16 WMMA, fp32 online softmax.
// Block = 128 threads (4 waves). grid = (S/64, B*H). Wave owns 16 query rows.
// All fragments load as 2x b128 (A-layout K-offsets per hi-half: two runs of 8, at 8*hi and 16+8*hi).
// ---------------------------------------------------------------------------
__global__ __launch_bounds__(128)
void k_flash_attn(const bf16_t* __restrict__ Q, const bf16_t* __restrict__ K,
                  const bf16_t* __restrict__ Vt, bf16_t* __restrict__ O, int S) {
    __shared__ bf16_t Ps[4][16][48];       // per-wave P restage (C-layout -> A-layout)

    const int wave = threadIdx.x >> 5;
    const int lane = threadIdx.x & 31;
    const int hi   = lane >> 4;
    const int ln   = lane & 15;
    const int bh   = blockIdx.y;
    const int b    = bh / NHEADS;
    const int h    = bh % NHEADS;
    const int q0   = blockIdx.x * 64 + wave * 16;

    const size_t base = (size_t)bh * S * HDIM;
    const bf16_t* Qp = Q  + base;
    const bf16_t* Kp = K  + base;
    const bf16_t* Vp = Vt + base;          // [HDIM][S]

    // Preload Q fragments: 4 feature chunks of 32
    AFrag qf[4];
#pragma unroll
    for (int c = 0; c < 4; ++c) {
        const bf16_t* qrow = Qp + (size_t)(q0 + ln) * HDIM + c * 32 + 8 * hi;
        *reinterpret_cast<int4*>(&qf[c].u[0]) = *reinterpret_cast<const int4*>(qrow);
        *reinterpret_cast<int4*>(&qf[c].u[4]) = *reinterpret_cast<const int4*>(qrow + 16);
    }

    float mrow[8], lrow[8];
    v8f oacc[8];
    const v8f vzero = {};
#pragma unroll
    for (int r = 0; r < 8; ++r) { mrow[r] = -3.0e38f; lrow[r] = 0.f; }
#pragma unroll
    for (int c = 0; c < 8; ++c) oacc[c] = vzero;

    const float scale = 0.08838834764831845f;   // 1/sqrt(128)
    const int   qmax  = q0 + 15;

    for (int kt = 0; kt <= qmax; kt += 32) {
        // ---- scores S = Q K^T for a 16x32 key strip ----
        v8f s0 = {}, s1 = {};
#pragma unroll
        for (int c = 0; c < 4; ++c) {
            AFrag b0, b1;
            const bf16_t* kr0 = Kp + (size_t)(kt + ln)      * HDIM + c * 32 + 8 * hi;
            const bf16_t* kr1 = Kp + (size_t)(kt + 16 + ln) * HDIM + c * 32 + 8 * hi;
            *reinterpret_cast<int4*>(&b0.u[0]) = *reinterpret_cast<const int4*>(kr0);
            *reinterpret_cast<int4*>(&b0.u[4]) = *reinterpret_cast<const int4*>(kr0 + 16);
            *reinterpret_cast<int4*>(&b1.u[0]) = *reinterpret_cast<const int4*>(kr1);
            *reinterpret_cast<int4*>(&b1.u[4]) = *reinterpret_cast<const int4*>(kr1 + 16);
            s0 = __builtin_amdgcn_wmma_f32_16x16x32_bf16(false, qf[c].v, false, b0.v,
                                                         (short)0, s0, false, false);
            s1 = __builtin_amdgcn_wmma_f32_16x16x32_bf16(false, qf[c].v, false, b1.v,
                                                         (short)0, s1, false, false);
        }

        // ---- causal mask + online softmax (row = r + 8*hi, col = ln) ----
        float p0[8], p1[8];
#pragma unroll
        for (int r = 0; r < 8; ++r) {
            const int qg = q0 + 8 * hi + r;
            float v0 = s0[r] * scale;
            float v1 = s1[r] * scale;
            if (kt + ln > qg)      v0 = -3.0e38f;
            if (kt + 16 + ln > qg) v1 = -3.0e38f;
            float mx = fmaxf(v0, v1);
#pragma unroll
            for (int off = 1; off < 16; off <<= 1)
                mx = fmaxf(mx, __shfl_xor(mx, off, 32));
            const float mnew  = fmaxf(mrow[r], mx);
            const float alpha = __expf(mrow[r] - mnew);
            mrow[r] = mnew;
            v0 = __expf(v0 - mnew);
            v1 = __expf(v1 - mnew);
            float sum = v0 + v1;
#pragma unroll
            for (int off = 1; off < 16; off <<= 1)
                sum += __shfl_xor(sum, off, 32);
            lrow[r] = lrow[r] * alpha + sum;
            p0[r] = v0; p1[r] = v1;
#pragma unroll
            for (int c = 0; c < 8; ++c) oacc[c][r] *= alpha;
        }

        // ---- restage P through LDS (C-layout -> A-layout) ----
#pragma unroll
        for (int r = 0; r < 8; ++r) {
            Ps[wave][8 * hi + r][ln]      = to_bf16(p0[r]);
            Ps[wave][8 * hi + r][16 + ln] = to_bf16(p1[r]);
        }
        AFrag pf;
        {
            const bf16_t* prow = &Ps[wave][ln][8 * hi];
            *reinterpret_cast<int4*>(&pf.u[0]) = *reinterpret_cast<const int4*>(prow);
            *reinterpret_cast<int4*>(&pf.u[4]) = *reinterpret_cast<const int4*>(prow + 16);
        }

        // ---- O += P V  (Vt rows = features, contiguous keys) ----
#pragma unroll
        for (int c = 0; c < 8; ++c) {
            AFrag vf;
            const bf16_t* vrow = Vp + (size_t)(c * 16 + ln) * S + kt + 8 * hi;
            *reinterpret_cast<int4*>(&vf.u[0]) = *reinterpret_cast<const int4*>(vrow);
            *reinterpret_cast<int4*>(&vf.u[4]) = *reinterpret_cast<const int4*>(vrow + 16);
            oacc[c] = __builtin_amdgcn_wmma_f32_16x16x32_bf16(false, pf.v, false, vf.v,
                                                              (short)0, oacc[c], false, false);
        }
    }

    // ---- normalize and write O as [B,S,D] bf16 ----
#pragma unroll
    for (int r = 0; r < 8; ++r) {
        const float inv = 1.0f / lrow[r];
        const size_t row = (size_t)b * S + (q0 + 8 * hi + r);
        bf16_t* orow = O + row * (size_t)DIMC + h * HDIM;
#pragma unroll
        for (int c = 0; c < 8; ++c)
            orow[c * 16 + ln] = to_bf16(oacc[c][r] * inv);
    }
}

// ---------------------------------------------------------------------------
// fp32 elementwise epilogues
// ---------------------------------------------------------------------------
__global__ __launch_bounds__(256)
void k_add_f32(const float* __restrict__ a, const float* __restrict__ b,
               float* __restrict__ c, size_t n) {
    size_t i = (size_t)blockIdx.x * blockDim.x + threadIdx.x;
    if (i < n) c[i] = a[i] + b[i];
}

__global__ __launch_bounds__(256)
void k_gelu_bf16(const float* __restrict__ in, bf16_t* __restrict__ out, size_t n) {
    size_t i = (size_t)blockIdx.x * blockDim.x + threadIdx.x;
    if (i >= n) return;
    const float x = in[i];
    const float t = tanhf(0.7978845608028654f * (x + 0.044715f * x * x * x));
    out[i] = to_bf16(0.5f * x * (1.0f + t));
}

// ---------------------------------------------------------------------------
// Orchestration
// ---------------------------------------------------------------------------
extern "C" void kernel_launch(void* const* d_in, const int* in_sizes, int n_in,
                              void* d_out, int out_size, void* d_ws, size_t ws_size,
                              hipStream_t stream) {
    (void)in_sizes; (void)n_in; (void)out_size; (void)ws_size;

    const float* x      = (const float*)d_in[0];
    const float* w_qkv  = (const float*)d_in[1];
    const float* w_out  = (const float*)d_in[2];
    const float* w_fc   = (const float*)d_in[3];
    const float* w_proj = (const float*)d_in[4];
    const float* g_in   = (const float*)d_in[5];
    const float* g_ff   = (const float*)d_in[6];
    float* out = (float*)d_out;

    constexpr int B = 2, S = 2048, D = DIMC, N3 = 3 * DIMC, FF = FFDIM;
    constexpr int M = B * S;

    char* p = (char*)d_ws;
    auto alloc = [&](size_t bytes) -> void* {
        void* r = (void*)p;
        p += (bytes + 255) & ~(size_t)255;
        return r;
    };

    bf16_t* wqkvT  = (bf16_t*)alloc((size_t)N3 * D * 2);
    bf16_t* woutT  = (bf16_t*)alloc((size_t)D * D * 2);
    bf16_t* wfcT   = (bf16_t*)alloc((size_t)FF * D * 2);
    bf16_t* wprojT = (bf16_t*)alloc((size_t)D * FF * 2);
    bf16_t* hb     = (bf16_t*)alloc((size_t)M * D * 2);          // rmsnorm out (reused)
    bf16_t* qb     = (bf16_t*)alloc((size_t)M * D * 2);          // [B,H,S,HD]
    bf16_t* kb     = (bf16_t*)alloc((size_t)M * D * 2);
    bf16_t* vtb    = (bf16_t*)alloc((size_t)M * D * 2);          // [B,H,HD,S]
    bf16_t* attnb  = (bf16_t*)alloc((size_t)M * D * 2);          // attention out bf16
    float*  x1     = (float*) alloc((size_t)M * D * 4);          // residual 1
    float*  big    = (float*) alloc((size_t)M * N3 * 4);         // qkv / attn_proj / ff / proj
    bf16_t* gelub  = (bf16_t*)alloc((size_t)M * FF * 2);

    // 1. weights -> bf16, transposed to N x K
    k_transpose_cast<<<dim3(N3 / 32, D / 32), 256, 0, stream>>>(w_qkv,  wqkvT,  D,  N3);
    k_transpose_cast<<<dim3(D  / 32, D / 32), 256, 0, stream>>>(w_out,  woutT,  D,  D);
    k_transpose_cast<<<dim3(FF / 32, D / 32), 256, 0, stream>>>(w_fc,   wfcT,   D,  FF);
    k_transpose_cast<<<dim3(D  / 32, FF / 32), 256, 0, stream>>>(w_proj, wprojT, FF, D);

    // 2. h = rmsnorm(x, g_in)  (bf16)
    k_rmsnorm_bf16<<<M, 256, 0, stream>>>(x, g_in, hb, D);

    // 3. qkv = h @ w_qkv  (fp32 out)
    k_gemm_bf16<<<dim3(N3 / GBN, M / GBM), 256, 0, stream>>>(hb, wqkvT, big, M, N3, D);

    // 4. RoPE + head split (+ V transpose)
    {
        size_t total = (size_t)B * S * NHEADS * (HDIM / 2);
        k_rope_split<<<(unsigned)((total + 255) / 256), 256, 0, stream>>>(big, qb, kb, vtb, B, S);
    }

    // 5. causal flash attention
    k_flash_attn<<<dim3(S / 64, B * NHEADS), 128, 0, stream>>>(qb, kb, vtb, attnb, S);

    // 6. attn_proj = attn @ w_out ; x1 = x + attn_proj
    k_gemm_bf16<<<dim3(D / GBN, M / GBM), 256, 0, stream>>>(attnb, woutT, big, M, D, D);
    k_add_f32<<<(unsigned)(((size_t)M * D + 255) / 256), 256, 0, stream>>>(x, big, x1, (size_t)M * D);

    // 7. h2 = rmsnorm(x1, g_ff)
    k_rmsnorm_bf16<<<M, 256, 0, stream>>>(x1, g_ff, hb, D);

    // 8. ff = h2 @ w_fc ; gelu
    k_gemm_bf16<<<dim3(FF / GBN, M / GBM), 256, 0, stream>>>(hb, wfcT, big, M, FF, D);
    k_gelu_bf16<<<(unsigned)(((size_t)M * FF + 255) / 256), 256, 0, stream>>>(big, gelub, (size_t)M * FF);

    // 9. proj = gelu @ w_proj ; out = x1 + proj
    k_gemm_bf16<<<dim3(D / GBN, M / GBM), 256, 0, stream>>>(gelub, wprojT, big, M, D, FF);
    k_add_f32<<<(unsigned)(((size_t)M * D + 255) / 256), 256, 0, stream>>>(x1, big, out, (size_t)M * D);
}
